// TransformerDecoderLayer_47897475285076
// MI455X (gfx1250) — matrix operand order repaired and checked by reading, not compiled, not run on hardware
//
#include <hip/hip_runtime.h>
#include <stdint.h>

// ---------------------------------------------------------------------------
// Problem constants: B=8192, S=4, E=1024, H=16, FF4=4096, HD=64, BS=32768
// ---------------------------------------------------------------------------

#define USE_ASYNC_LDS 1   // global_load_async_to_lds_b128
#define USE_TDM       1   // probe: tensor_load_to_lds + s_wait_tensorcnt

typedef __attribute__((ext_vector_type(16))) __bf16 v16bf;
typedef __attribute__((ext_vector_type(8)))  float  v8f;

union Frag32B { uint4 q[2]; v16bf v; };

__device__ __forceinline__ unsigned short f2bf(float f) {
  union { float f; unsigned u; } x; x.f = f;
  unsigned r = x.u + 0x7FFFu + ((x.u >> 16) & 1u);
  return (unsigned short)(r >> 16);
}
__device__ __forceinline__ float bf2f(unsigned short h) {
  union { unsigned u; float f; } x; x.u = ((unsigned)h) << 16;
  return x.f;
}

__device__ __forceinline__ v8f wmma_bf16(const Frag32B& a, const Frag32B& b, v8f c) {
  return __builtin_amdgcn_wmma_f32_16x16x32_bf16(
      false, a.v, false, b.v, (short)0, c, false, false);
}

__device__ __forceinline__ void g2l_16B(void* lds, const void* g) {
#if USE_ASYNC_LDS
  unsigned loff = (unsigned)(uintptr_t)lds;
  asm volatile("global_load_async_to_lds_b128 %0, %1, off"
               :: "v"(loff), "v"(g) : "memory");
#else
  *(uint4*)lds = *(const uint4*)g;
#endif
}
__device__ __forceinline__ void g2l_wait() {
#if USE_ASYNC_LDS
  asm volatile("s_wait_asynccnt 0" ::: "memory");
#endif
}

#if USE_TDM
typedef __attribute__((ext_vector_type(4))) unsigned tdm_g0_t;
typedef __attribute__((ext_vector_type(8))) unsigned tdm_g1_t;
// 1-D linear tile: elems8B consecutive 8-byte elements, global -> LDS.
__device__ __forceinline__ void tdm_load_1d(unsigned ldsOff, const void* gptr,
                                            unsigned elems8B) {
  unsigned long long ga = (unsigned long long)(uintptr_t)gptr;
  tdm_g0_t g0 = {
    1u,                                               // count=1 (valid), user D#
    ldsOff,                                           // lds_addr
    (unsigned)ga,                                     // global_addr[31:0]
    (unsigned)((ga >> 32) & 0x1FFFFFFu) | (2u << 30)  // addr[56:32], type=2
  };
  tdm_g1_t g1 = {
    3u << 16,                                  // data_size=8B; mask=0
    (elems8B & 0xFFFFu) << 16,                 // tensor_dim0[15:0]
    (elems8B >> 16) | (1u << 16),              // tensor_dim0[31:16], tensor_dim1=1
    (elems8B & 0xFFFFu) << 16,                 // tile_dim0
    0u,                                        // tile_dim1=0, tile_dim2=0
    elems8B,                                   // tensor_dim0_stride[31:0]
    0u, 0u
  };
  asm volatile("tensor_load_to_lds %0, %1" :: "s"(g0), "s"(g1) : "memory");
}
#endif

// A-frag (16x32 bf16): lane = row M; halves 0-7 = K[(lane/16)*8..], 8-15 = K[16+..]
__device__ __forceinline__ void load_a_frag(Frag32B& f, const unsigned short* rowBase, int lane) {
  int kh = (lane >> 4) * 8;
  f.q[0] = *(const uint4*)(rowBase + kh);
  f.q[1] = *(const uint4*)(rowBase + 16 + kh);
}
// B-frag (32x16 bf16): lane = col N; halves 0-7 = K[(lane/16)*8..], 8-15 = K[16+..]
__device__ __forceinline__ void load_b_frag(Frag32B& f, const unsigned short* rowBase, int lane) {
  int kh = (lane >> 4) * 8;
  f.q[0] = *(const uint4*)(rowBase + kh);
  f.q[1] = *(const uint4*)(rowBase + 16 + kh);
}

// ---------------------------------------------------------------------------
__global__ void k_cvt(const float* __restrict__ src, unsigned short* __restrict__ dst, int n) {
  int i = blockIdx.x * blockDim.x + threadIdx.x;
  if (i < n) dst[i] = f2bf(src[i]);
}

__global__ void k_init_out(const float* __restrict__ b3, float* __restrict__ out) {
  int i = blockIdx.x * blockDim.x + threadIdx.x;
  if (i < 8192 * 4) out[i] = b3[i & 3];
}

__global__ void k_embed(const int* __restrict__ cidx, const int* __restrict__ vidx,
                        const float* __restrict__ pos, const float* __restrict__ cemb,
                        const float* __restrict__ vemb, unsigned short* __restrict__ xbf) {
  long long i = (long long)blockIdx.x * blockDim.x + threadIdx.x;
  int row = (int)(i >> 10);
  int e   = (int)(i & 1023);
  int s   = row & 3;
  float v = (e < 512) ? cemb[cidx[row] * 512 + e]
                      : vemb[vidx[row] * 512 + (e - 512)];
  v += pos[s * 1024 + e];
  xbf[i] = f2bf(v);
}

// ---------------------------------------------------------------------------
// Shared 128x128 GEMM machinery (K step 32, double-buffered async staging).
// 8 waves: wave grid 4(M) x 2(N); wave tile 32x64 => 8 WMMA per K-step.

__device__ __forceinline__ void stage_tile(
    const unsigned short* __restrict__ A, const unsigned short* __restrict__ W,
    unsigned short* As, unsigned short* Bs, int m0, int n0, int k0,
    int t, int lda, int ldw) {
  const int r = t >> 1, seg = t & 1;
  const unsigned short* ga = A + (size_t)(m0 + r) * lda + k0 + seg * 16;
  g2l_16B(&As[r * 32 + seg * 16],     ga);
  g2l_16B(&As[r * 32 + seg * 16 + 8], ga + 8);
  const unsigned short* gb = W + (size_t)(n0 + r) * ldw + k0 + seg * 16;
  g2l_16B(&Bs[r * 32 + seg * 16],     gb);
  g2l_16B(&Bs[r * 32 + seg * 16 + 8], gb + 8);
}

__device__ __forceinline__ void compute_step(
    const unsigned short* As, const unsigned short* Bs,
    int lane, int wm, int wn, v8f acc[2][4]) {
  const int fr = lane & 15;
  Frag32B a[2], b[4];
#pragma unroll
  for (int i = 0; i < 2; ++i) load_a_frag(a[i], &As[(wm + i * 16 + fr) * 32], lane);
#pragma unroll
  for (int j = 0; j < 4; ++j) load_b_frag(b[j], &Bs[(wn + j * 16 + fr) * 32], lane);
#pragma unroll
  for (int i = 0; i < 2; ++i)
#pragma unroll
    for (int j = 0; j < 4; ++j) acc[i][j] = wmma_bf16(a[i], b[j], acc[i][j]);
}

// QKV: qkv[32768,3072] = x_bf @ Wqkv^T + b
__global__ __launch_bounds__(256) void k_gemm_qkv(
    const unsigned short* __restrict__ A, const unsigned short* __restrict__ W,
    const float* __restrict__ bias, unsigned short* __restrict__ C) {
  __shared__ unsigned short As[2][128 * 32];
  __shared__ unsigned short Bs[2][128 * 32];
  const int t = threadIdx.x, lane = t & 31, wid = t >> 5;
  const int m0 = blockIdx.y * 128, n0 = blockIdx.x * 128;
  const int wm = (wid >> 1) * 32, wn = (wid & 1) * 64;
  const v8f vz = {0.f,0.f,0.f,0.f,0.f,0.f,0.f,0.f};
  v8f acc[2][4] = {{vz,vz,vz,vz},{vz,vz,vz,vz}};

  stage_tile(A, W, As[0], Bs[0], m0, n0, 0, t, 1024, 1024);
  g2l_wait();
  __syncthreads();
  const int NIT = 1024 / 32;
  for (int it = 0; it < NIT; ++it) {
    const int buf = it & 1;
    if (it + 1 < NIT)
      stage_tile(A, W, As[buf ^ 1], Bs[buf ^ 1], m0, n0, (it + 1) * 32, t, 1024, 1024);
    compute_step(As[buf], Bs[buf], lane, wm, wn, acc);
    g2l_wait();          // stage it+1 landed (my wave)
    __syncthreads();     // all waves: landed + done reading buf
  }

  const int rb = (lane >> 4) * 8, c = lane & 15;
#pragma unroll
  for (int i = 0; i < 2; ++i)
    for (int j = 0; j < 4; ++j)
      for (int q = 0; q < 8; ++q) {
        int mg = m0 + wm + i * 16 + rb + q;
        int ng = n0 + wn + j * 16 + c;
        C[(size_t)mg * 3072 + ng] = f2bf(acc[i][j][q] + bias[ng]);
      }
}

// out-proj (M=8192 last-position rows): x1 = x[row*4+3] + o3 @ Wo^T + b
__global__ __launch_bounds__(256) void k_gemm_outproj(
    const unsigned short* __restrict__ A, const unsigned short* __restrict__ W,
    const float* __restrict__ bias, const unsigned short* __restrict__ Xres,
    unsigned short* __restrict__ C) {
  __shared__ unsigned short As[2][128 * 32];
  __shared__ unsigned short Bs[2][128 * 32];
  const int t = threadIdx.x, lane = t & 31, wid = t >> 5;
  const int m0 = blockIdx.y * 128, n0 = blockIdx.x * 128;
  const int wm = (wid >> 1) * 32, wn = (wid & 1) * 64;
  const v8f vz = {0.f,0.f,0.f,0.f,0.f,0.f,0.f,0.f};
  v8f acc[2][4] = {{vz,vz,vz,vz},{vz,vz,vz,vz}};

  stage_tile(A, W, As[0], Bs[0], m0, n0, 0, t, 1024, 1024);
  g2l_wait();
  __syncthreads();
  const int NIT = 1024 / 32;
  for (int it = 0; it < NIT; ++it) {
    const int buf = it & 1;
    if (it + 1 < NIT)
      stage_tile(A, W, As[buf ^ 1], Bs[buf ^ 1], m0, n0, (it + 1) * 32, t, 1024, 1024);
    compute_step(As[buf], Bs[buf], lane, wm, wn, acc);
    g2l_wait();
    __syncthreads();
  }

  const int rb = (lane >> 4) * 8, c = lane & 15;
#pragma unroll
  for (int i = 0; i < 2; ++i)
    for (int j = 0; j < 4; ++j)
      for (int q = 0; q < 8; ++q) {
        int mg = m0 + wm + i * 16 + rb + q;
        int ng = n0 + wn + j * 16 + c;
        float r = bf2f(Xres[((size_t)mg * 4 + 3) * 1024 + ng]);
        C[(size_t)mg * 1024 + ng] = f2bf(acc[i][j][q] + bias[ng] + r);
      }
}

// ---------------------------------------------------------------------------
// Attention, last position only (row 3 attends all 4 keys; no mask needed).
__global__ void k_attn(const unsigned short* __restrict__ qkv,
                       unsigned short* __restrict__ o3) {
  int i = blockIdx.x * blockDim.x + threadIdx.x;
  int b = i >> 4, h = i & 15;
  const unsigned short* base = qkv + (size_t)b * 4 * 3072;
  float qv[64];
#pragma unroll
  for (int d = 0; d < 64; ++d) qv[d] = bf2f(base[3 * 3072 + h * 64 + d]);
  float sc[4], mx = -1e30f;
  for (int kk = 0; kk < 4; ++kk) {
    float s = 0.f;
    for (int d = 0; d < 64; ++d)
      s += qv[d] * bf2f(base[kk * 3072 + 1024 + h * 64 + d]);
    sc[kk] = s * 0.125f;
    mx = fmaxf(mx, sc[kk]);
  }
  float den = 0.f;
  for (int kk = 0; kk < 4; ++kk) { sc[kk] = __expf(sc[kk] - mx); den += sc[kk]; }
  float inv = 1.f / den;
  for (int d = 0; d < 64; ++d) {
    float o = 0.f;
    for (int kk = 0; kk < 4; ++kk)
      o += sc[kk] * bf2f(base[kk * 3072 + 2048 + h * 64 + d]);
    o3[(size_t)b * 1024 + h * 64 + d] = f2bf(o * inv);
  }
}

// ---------------------------------------------------------------------------
// Fused FFN + final projection. 16 rows/block; h[16,4096] bf16 lives in LDS.
// x1 block staged by the Tensor Data Mover (1-D 32KB tile).
__global__ __launch_bounds__(256) void k_ffn(
    const unsigned short* __restrict__ X1, const unsigned short* __restrict__ W1,
    const unsigned short* __restrict__ W2,
    const float* __restrict__ b1, const float* __restrict__ b2,
    const float* __restrict__ w3, float* __restrict__ out) {
  extern __shared__ unsigned short smem[];
  unsigned short* xs = smem;              // 16*1024
  unsigned short* hs = smem + 16 * 1024;  // 16*4096
  const int t = threadIdx.x, lane = t & 31, wid = t >> 5;
  const int m0 = blockIdx.x * 16;
  const int fr = lane & 15;
  const v8f vz = {0.f,0.f,0.f,0.f,0.f,0.f,0.f,0.f};

#if USE_TDM
  if (wid == 0) {
    tdm_load_1d((unsigned)(uintptr_t)xs, X1 + (size_t)m0 * 1024,
                (16u * 1024u * 2u) / 8u);          // 4096 x 8B
    __builtin_amdgcn_s_wait_tensorcnt(0);
  }
#else
  {
    const uint4* src = (const uint4*)(X1 + (size_t)m0 * 1024);
    uint4* dst = (uint4*)xs;
#pragma unroll
    for (int u = 0; u < 8; ++u) dst[t * 8 + u] = src[t * 8 + u];
  }
#endif
  __syncthreads();

  // ---- Phase 1: h = relu(x1 @ w1^T + b1) -> LDS, 2 N-subtiles per wave ----
  for (int n0 = 0; n0 < 4096; n0 += 256) {
    const int nc0 = n0 + wid * 32;
    v8f acc[2] = {vz, vz};
    for (int k0 = 0; k0 < 1024; k0 += 32) {
      Frag32B a, b0, b1f;
      load_a_frag(a, &xs[fr * 1024 + k0], lane);
      load_b_frag(b0,  W1 + (size_t)(nc0 + fr) * 1024 + k0, lane);
      load_b_frag(b1f, W1 + (size_t)(nc0 + 16 + fr) * 1024 + k0, lane);
      acc[0] = wmma_bf16(a, b0,  acc[0]);
      acc[1] = wmma_bf16(a, b1f, acc[1]);
    }
    const int rb = (lane >> 4) * 8, c = lane & 15;
#pragma unroll
    for (int j = 0; j < 2; ++j) {
      const int nc = nc0 + j * 16 + c;
#pragma unroll
      for (int q = 0; q < 8; ++q) {
        float h = fmaxf(acc[j][q] + b1[nc], 0.f);
        hs[(rb + q) * 4096 + nc] = f2bf(h);
      }
    }
  }
  __syncthreads();

  // ---- Phase 2: y = x1 + h @ w2^T + b2, fused w3 projection ----
  for (int n0 = 0; n0 < 1024; n0 += 256) {
    const int nc0 = n0 + wid * 32;
    v8f acc[2] = {vz, vz};
    for (int k0 = 0; k0 < 4096; k0 += 32) {
      Frag32B a, b0, b1f;
      load_a_frag(a, &hs[fr * 4096 + k0], lane);
      load_b_frag(b0,  W2 + (size_t)(nc0 + fr) * 4096 + k0, lane);
      load_b_frag(b1f, W2 + (size_t)(nc0 + 16 + fr) * 4096 + k0, lane);
      acc[0] = wmma_bf16(a, b0,  acc[0]);
      acc[1] = wmma_bf16(a, b1f, acc[1]);
    }
    const int rb = (lane >> 4) * 8, c = lane & 15;
#pragma unroll
    for (int j = 0; j < 2; ++j) {
      const int nc = nc0 + j * 16 + c;
#pragma unroll
      for (int q = 0; q < 8; ++q) {
        const int r = rb + q, mg = m0 + r;
        float y = acc[j][q] + b2[nc] + bf2f(xs[r * 1024 + nc]);
#pragma unroll
        for (int cc = 0; cc < 4; ++cc) {
          float p = y * w3[cc * 1024 + nc];
          p += __shfl_xor(p, 1);
          p += __shfl_xor(p, 2);
          p += __shfl_xor(p, 4);
          p += __shfl_xor(p, 8);
          if ((lane & 15) == 0) atomicAdd(&out[mg * 4 + cc], p);
        }
      }
    }
  }
}

// ---------------------------------------------------------------------------
extern "C" void kernel_launch(void* const* d_in, const int* in_sizes, int n_in,
                              void* d_out, int out_size, void* d_ws, size_t ws_size,
                              hipStream_t stream) {
  const int*   choice_idx = (const int*)d_in[0];
  const int*   val_idx    = (const int*)d_in[1];
  const float* pos_emb    = (const float*)d_in[2];
  const float* choice_emb = (const float*)d_in[3];
  const float* val_emb    = (const float*)d_in[4];
  const float* in_proj_w  = (const float*)d_in[5];
  const float* in_proj_b  = (const float*)d_in[6];
  const float* out_proj_w = (const float*)d_in[7];
  const float* out_proj_b = (const float*)d_in[8];
  const float* w1 = (const float*)d_in[9];
  const float* b1 = (const float*)d_in[10];
  const float* w2 = (const float*)d_in[11];
  const float* b2 = (const float*)d_in[12];
  const float* w3 = (const float*)d_in[13];
  const float* b3 = (const float*)d_in[14];
  float* out = (float*)d_out;

  unsigned short* ws = (unsigned short*)d_ws;
  size_t off = 0;
  unsigned short* wqkv = ws + off; off += 3072ull * 1024;
  unsigned short* wout = ws + off; off += 1024ull * 1024;
  unsigned short* w1b  = ws + off; off += 4096ull * 1024;
  unsigned short* w2b  = ws + off; off += 1024ull * 4096;
  unsigned short* xbf  = ws + off; off += 32768ull * 1024;
  unsigned short* qkvb = ws + off; off += 32768ull * 3072;
  unsigned short* o3b  = ws + off; off += 8192ull * 1024;
  unsigned short* x1b  = ws + off; off += 8192ull * 1024;

  k_cvt<<<(3072 * 1024 + 255) / 256, 256, 0, stream>>>(in_proj_w,  wqkv, 3072 * 1024);
  k_cvt<<<(1024 * 1024 + 255) / 256, 256, 0, stream>>>(out_proj_w, wout, 1024 * 1024);
  k_cvt<<<(4096 * 1024 + 255) / 256, 256, 0, stream>>>(w1, w1b, 4096 * 1024);
  k_cvt<<<(1024 * 4096 + 255) / 256, 256, 0, stream>>>(w2, w2b, 1024 * 4096);
  k_init_out<<<128, 256, 0, stream>>>(b3, out);

  k_embed<<<131072, 256, 0, stream>>>(choice_idx, val_idx, pos_emb,
                                      choice_emb, val_emb, xbf);

  dim3 gq(3072 / 128, 32768 / 128);
  k_gemm_qkv<<<gq, 256, 0, stream>>>(xbf, wqkv, in_proj_b, qkvb);

  k_attn<<<(8192 * 16) / 256, 256, 0, stream>>>(qkvb, o3b);

  dim3 go(1024 / 128, 8192 / 128);
  k_gemm_outproj<<<go, 256, 0, stream>>>(o3b, wout, out_proj_b, xbf, x1b);

  const int ffn_smem = (16 * 1024 + 16 * 4096) * 2;   // 160 KB
  hipFuncSetAttribute(reinterpret_cast<const void*>(&k_ffn),
                      hipFuncAttributeMaxDynamicSharedMemorySize, ffn_smem);
  k_ffn<<<8192 / 16, 256, ffn_smem, stream>>>(x1b, w1b, w2b, b1, b2, w3, out);
}
